// AttentionLayer_47253230190929
// MI455X (gfx1250) — matrix operand order repaired and checked by reading, not compiled, not run on hardware
//
#include <hip/hip_runtime.h>

typedef __attribute__((ext_vector_type(16))) _Float16 v16h;
typedef __attribute__((ext_vector_type(8)))  float    v8f;

#define B_   64
#define T_   8192
#define H_   256
#define ROWS_PER_BLOCK 128
#define NBLK (T_ / ROWS_PER_BLOCK)   // 64 blocks along T per batch

// ---------------------------------------------------------------------------
// Kernel 1: q[b,:] = ht[b,:] @ W   (64 x 256 x 256 GEMM via WMMA f16->f32)
// One wave (32 threads) per 16x16 output tile. grid = (64/16)*(256/16) = 64.
// ---------------------------------------------------------------------------
__global__ __launch_bounds__(32)
void qgemm_kernel(const float* __restrict__ ht, const float* __restrict__ W,
                  float* __restrict__ q)
{
    const int lane = threadIdx.x & 31;
    const int l15  = lane & 15;
    const int kh   = lane >> 4;          // which half of the wave
    const int bt   = blockIdx.x >> 4;    // M tile (batches 16*bt .. +15)
    const int nt   = blockIdx.x & 15;    // N tile (output cols)
    const int ncol = nt * 16 + l15;

    const float* arow = ht + (size_t)(bt * 16 + l15) * H_;

    v8f c = {};
    #pragma unroll
    for (int kc = 0; kc < 8; ++kc) {
        const int k0 = kc * 32;
        // A fragment: 16x32 f16, ISA layout (K = kh*8 + i  /  16 + kh*8 + i)
        const float4 f0 = ((const float4*)(arow + k0 + kh * 8))[0];
        const float4 f1 = ((const float4*)(arow + k0 + kh * 8))[1];
        const float4 f2 = ((const float4*)(arow + k0 + 16 + kh * 8))[0];
        const float4 f3 = ((const float4*)(arow + k0 + 16 + kh * 8))[1];
        v16h a;
        a[0]=(_Float16)f0.x; a[1]=(_Float16)f0.y; a[2]=(_Float16)f0.z; a[3]=(_Float16)f0.w;
        a[4]=(_Float16)f1.x; a[5]=(_Float16)f1.y; a[6]=(_Float16)f1.z; a[7]=(_Float16)f1.w;
        a[8]=(_Float16)f2.x; a[9]=(_Float16)f2.y; a[10]=(_Float16)f2.z; a[11]=(_Float16)f2.w;
        a[12]=(_Float16)f3.x; a[13]=(_Float16)f3.y; a[14]=(_Float16)f3.z; a[15]=(_Float16)f3.w;
        // B fragment: 32x16 f16, same K layout as A (K = kh*8 + i  /  16 + kh*8 + i)
        v16h bm;
        #pragma unroll
        for (int i = 0; i < 16; ++i)
            bm[i] = (_Float16)W[(size_t)(k0 + kh * 8 + i + (i >= 8 ? 8 : 0)) * H_ + ncol];
        c = __builtin_amdgcn_wmma_f32_16x16x32_f16(
                false, a, false, bm, (short)0, c, false, false);
    }
    // D: lanes 0-15 -> M=r, lanes 16-31 -> M=8+r
    #pragma unroll
    for (int r = 0; r < 8; ++r)
        q[(size_t)(bt * 16 + 8 * kh + r) * H_ + ncol] = c[r];
}

// ---------------------------------------------------------------------------
// Kernel 2: fused scores + online softmax partials + weighted-sum partials.
// grid = (NBLK, B); 256 threads = 8 waves; each wave owns 16 rows of T.
// Scores via WMMA: S(16x16) = A(16x32 of h_0_t) x B(q replicated per column).
// Single HBM pass over h_0_t (second read is cache-hot).
// ---------------------------------------------------------------------------
__global__ __launch_bounds__(256)
void flash_kernel(const float* __restrict__ h0, const float* __restrict__ q,
                  float* __restrict__ pm, float* __restrict__ psum,
                  float* __restrict__ pacc)
{
    __shared__ float lds_acc[8][H_];
    __shared__ float lds_m[8];
    __shared__ float lds_s[8];

    const int tid  = threadIdx.x;
    const int wave = tid >> 5;
    const int lane = tid & 31;
    const int l15  = lane & 15;
    const int kh   = lane >> 4;
    const int b    = blockIdx.y;
    const int blk  = blockIdx.x;
    const int rowbase = blk * ROWS_PER_BLOCK + wave * 16;

    const float* h0b  = h0 + ((size_t)b * T_ + rowbase) * H_;   // 16 rows x 256
    const float* qb   = q  + (size_t)b * H_;
    const float* arow = h0b + (size_t)l15 * H_;

    v8f c = {};
    #pragma unroll
    for (int kc = 0; kc < 8; ++kc) {
        const int k0 = kc * 32;
        __builtin_prefetch(arow + k0 + 32, 0, 0);   // global_prefetch_b8, speculative
        const float4 f0 = ((const float4*)(arow + k0 + kh * 8))[0];
        const float4 f1 = ((const float4*)(arow + k0 + kh * 8))[1];
        const float4 f2 = ((const float4*)(arow + k0 + 16 + kh * 8))[0];
        const float4 f3 = ((const float4*)(arow + k0 + 16 + kh * 8))[1];
        v16h a;
        a[0]=(_Float16)f0.x; a[1]=(_Float16)f0.y; a[2]=(_Float16)f0.z; a[3]=(_Float16)f0.w;
        a[4]=(_Float16)f1.x; a[5]=(_Float16)f1.y; a[6]=(_Float16)f1.z; a[7]=(_Float16)f1.w;
        a[8]=(_Float16)f2.x; a[9]=(_Float16)f2.y; a[10]=(_Float16)f2.z; a[11]=(_Float16)f2.w;
        a[12]=(_Float16)f3.x; a[13]=(_Float16)f3.y; a[14]=(_Float16)f3.z; a[15]=(_Float16)f3.w;
        // B = q broadcast into every column: every D column equals the scores
        const float4 q0 = ((const float4*)(qb + k0 + kh * 8))[0];
        const float4 q1 = ((const float4*)(qb + k0 + kh * 8))[1];
        const float4 q2 = ((const float4*)(qb + k0 + 16 + kh * 8))[0];
        const float4 q3 = ((const float4*)(qb + k0 + 16 + kh * 8))[1];
        v16h bm;
        bm[0]=(_Float16)q0.x; bm[1]=(_Float16)q0.y; bm[2]=(_Float16)q0.z; bm[3]=(_Float16)q0.w;
        bm[4]=(_Float16)q1.x; bm[5]=(_Float16)q1.y; bm[6]=(_Float16)q1.z; bm[7]=(_Float16)q1.w;
        bm[8]=(_Float16)q2.x; bm[9]=(_Float16)q2.y; bm[10]=(_Float16)q2.z; bm[11]=(_Float16)q2.w;
        bm[12]=(_Float16)q3.x; bm[13]=(_Float16)q3.y; bm[14]=(_Float16)q3.z; bm[15]=(_Float16)q3.w;
        c = __builtin_amdgcn_wmma_f32_16x16x32_f16(
                false, a, false, bm, (short)0, c, false, false);
    }

    // c[r] = score of row (8*kh + r); identical across the N dimension.
    float mloc = c[0];
    #pragma unroll
    for (int r = 1; r < 8; ++r) mloc = fmaxf(mloc, c[r]);
    const float mw = fmaxf(mloc, __shfl_xor(mloc, 16, 32));   // max over all 16 rows

    float w[8], wo[8];
    float sums = 0.f;
    #pragma unroll
    for (int r = 0; r < 8; ++r) {
        w[r]  = __expf(c[r] - mw);          // rows 8*kh + r
        wo[r] = __shfl_xor(w[r], 16, 32);   // rows 8*(1-kh) + r
        sums += w[r] + wo[r];               // sum over all 16 (same in every lane)
    }

    // Weighted accumulation over this wave's 16 rows; lane owns cols lane+32*j.
    // Re-reads are L0/L2 resident (just streamed for the WMMA above).
    #pragma unroll
    for (int j = 0; j < 8; ++j) {
        const int col = lane + 32 * j;
        float s = 0.f;
        #pragma unroll
        for (int r = 0; r < 8; ++r) {
            s += w[r]  * h0b[(size_t)(8 * kh       + r) * H_ + col];
            s += wo[r] * h0b[(size_t)(8 * (1 - kh) + r) * H_ + col];
        }
        lds_acc[wave][col] = s;
    }
    if (lane == 0) { lds_m[wave] = mw; lds_s[wave] = sums; }
    __syncthreads();

    // Block-level combine of the 8 wave partials (online-softmax rescale).
    const int col = tid;                    // 256 threads -> one column each
    float Mb = lds_m[0];
    #pragma unroll
    for (int wv = 1; wv < 8; ++wv) Mb = fmaxf(Mb, lds_m[wv]);
    float total = 0.f, acc = 0.f;
    #pragma unroll
    for (int wv = 0; wv < 8; ++wv) {
        const float e = __expf(lds_m[wv] - Mb);
        total += e * lds_s[wv];
        acc   += e * lds_acc[wv][col];
    }
    pacc[((size_t)b * NBLK + blk) * H_ + col] = acc;
    if (tid == 0) { pm[b * NBLK + blk] = Mb; psum[b * NBLK + blk] = total; }
}

// ---------------------------------------------------------------------------
// Kernel 3: combine the NBLK per-block partials for each batch.
// grid = B, 256 threads (one per output column).
// ---------------------------------------------------------------------------
__global__ __launch_bounds__(256)
void reduce_kernel(const float* __restrict__ pm, const float* __restrict__ psum,
                   const float* __restrict__ pacc, float* __restrict__ out)
{
    const int b   = blockIdx.x;
    const int col = threadIdx.x;
    float M = pm[b * NBLK];
    for (int j = 1; j < NBLK; ++j) M = fmaxf(M, pm[b * NBLK + j]);
    float total = 0.f, acc = 0.f;
    for (int j = 0; j < NBLK; ++j) {
        const float e = __expf(pm[b * NBLK + j] - M);
        total += e * psum[b * NBLK + j];
        acc   += e * pacc[((size_t)b * NBLK + j) * H_ + col];
    }
    out[(size_t)b * H_ + col] = acc / total;
}

// ---------------------------------------------------------------------------
extern "C" void kernel_launch(void* const* d_in, const int* in_sizes, int n_in,
                              void* d_out, int out_size, void* d_ws, size_t ws_size,
                              hipStream_t stream)
{
    (void)in_sizes; (void)n_in; (void)out_size; (void)ws_size;
    const float* ht = (const float*)d_in[0];   // (B,1,H)
    const float* h0 = (const float*)d_in[1];   // (B,T,H)
    const float* W  = (const float*)d_in[2];   // (H,H)
    float* out = (float*)d_out;                // (B,1,H)

    float* q    = (float*)d_ws;                // B*H
    float* pm   = q    + (size_t)B_ * H_;      // B*NBLK
    float* psum = pm   + (size_t)B_ * NBLK;    // B*NBLK
    float* pacc = psum + (size_t)B_ * NBLK;    // B*NBLK*H  (~4 MB)

    qgemm_kernel<<<(B_ / 16) * (H_ / 16), 32, 0, stream>>>(ht, W, q);
    dim3 grid2(NBLK, B_);
    flash_kernel<<<grid2, 256, 0, stream>>>(h0, q, pm, psum, pacc);
    reduce_kernel<<<B_, 256, 0, stream>>>(pm, psum, pacc, out);
}
